// STGformer_80126909874210
// MI455X (gfx1250) — matrix operand order repaired and checked by reading, not compiled, not run on hardware
//
#include <hip/hip_runtime.h>
#include <hip/hip_bf16.h>

// ---------------- model constants ----------------
#define B_   32
#define T_   12
#define NV_  512
#define D_   160
#define H_   8
#define HD_  20
#define S_   (B_*T_*NV_)     // 196608 rows of (B,T,N)
#define BN_  (B_*NV_)        // 16384 rows of (B,N)
#define TD_  (T_*D_)         // 1920
#define HID_ (2*D_)          // 320

typedef __attribute__((ext_vector_type(16))) __bf16 v16bf;
typedef __attribute__((ext_vector_type(8)))  __bf16 v8bf;
typedef __attribute__((ext_vector_type(4)))  __bf16 v4bf;
typedef __attribute__((ext_vector_type(8)))  float  v8f;

__device__ __forceinline__ __bf16 f2bf(float f) {
  unsigned u = __builtin_bit_cast(unsigned, f);
  u += 0x7FFFu + ((u >> 16) & 1u);               // round-to-nearest-even
  unsigned short h = (unsigned short)(u >> 16);
  return __builtin_bit_cast(__bf16, h);
}

union V16U { v16bf v; v8bf h[2]; };

// ---------------------------------------------------------------------------
// Batched WMMA GEMM:  C[z] = A[z%aMod] * op(B[z]) (+bias)(+res)(relu)
//   A: (M,K) row-major.  B: transB ? (N,K) row-major : (K,N) row-major.
// K % 32 == 0 and M % 64 == 0 for every GEMM in this model (N may be ragged).
// Block = 128 threads (4 waves). Block tile = 64 rows x 64 cols.
// Each wave owns 16 rows x 64 cols: 1 A-fragment feeds 4 WMMAs per k-step.
// B tile (64n x 32k, staged K-contiguous per n-row) is shared by all 4 waves.
// Phased staging: issue all global_load_b128 -> one wait -> bf16 pack -> LDS;
// issue all fragment ds_load_b128 -> WMMA chain.  fp32 accumulate.
// ---------------------------------------------------------------------------
#define GW 4
__global__ __launch_bounds__(128) void k_gemm(
    const float* __restrict__ A, int lda, long aBatch, int aMod,
    const float* __restrict__ Bm, int ldb, long bBatch, int transB,
    const float* __restrict__ bias,
    const float* __restrict__ res,
    float* __restrict__ C, int ldc, long cBatch,
    int M, int N, int K, int relu)
{
  __shared__ __bf16 sA[GW][16 * 32];   // per-wave A tile  [m][k]
  __shared__ __bf16 sB[64 * 32];       // shared  B tile  [n][k] (K-contiguous)

  const int tid  = threadIdx.x;
  const int wave = tid >> 5;
  const int lane = tid & 31;
  const int z    = blockIdx.z;

  const float* Ab = A  + (long)(z % aMod) * aBatch;
  const float* Bb = Bm + (long)z * bBatch;
  float*       Cb = C  + (long)z * cBatch;
  const float* Rb = res ? res + (long)z * cBatch : nullptr;

  const int n0 = blockIdx.x * 64;
  const int m0 = blockIdx.y * 64 + wave * 16;

  v8f acc[4];
  #pragma unroll
  for (int t = 0; t < 4; ++t) acc[t] = (v8f){0.f,0.f,0.f,0.f,0.f,0.f,0.f,0.f};

  __bf16* pA = &sA[wave][0];

  for (int k0 = 0; k0 < K; k0 += 32) {
    // ================= phase 1: issue ALL global loads ====================
    float4 fa[4];
    #pragma unroll
    for (int it = 0; it < 4; ++it) {
      int i = it * 32 + lane;          // 0..127 (per wave)
      int r = i >> 3, c4 = (i & 7) * 4;
      fa[it] = *(const float4*)&Ab[(long)(m0 + r) * lda + k0 + c4];
    }
    float4 fb[4];
    float  bmsk[4];
    if (transB) {                      // weights (N,K): K-contiguous rows
      #pragma unroll
      for (int it = 0; it < 4; ++it) {
        int i = it * 128 + tid;        // 0..511 (block)
        int r = i >> 3, c4 = (i & 7) * 4;
        int col = n0 + r;
        int colc = col < N ? col : N - 1;
        bmsk[it] = col < N ? 1.f : 0.f;
        fb[it] = *(const float4*)&Bb[(long)colc * ldb + k0 + c4];
      }
    } else {                           // plain (K,N)
      #pragma unroll
      for (int it = 0; it < 4; ++it) {
        int i = it * 128 + tid;
        int kr = i >> 4, c4 = (i & 15) * 4;
        fb[it] = *(const float4*)&Bb[(long)(k0 + kr) * ldb + n0 + c4];
        bmsk[it] = 1.f;                // N-edge masked at store below
      }
    }
    if (k0 + 32 < K) {
      __builtin_prefetch(&Ab[(long)(m0 + (lane & 15)) * lda + k0 + 32], 0, 1);
      __builtin_prefetch(&Bb[(long)((transB ? (n0 + (lane & 15)) : k0 + 32)) * ldb +
                             (transB ? (k0 + 32) : n0)], 0, 1);
    }

    // ================= phase 2: pack bf16 -> LDS ==========================
    #pragma unroll
    for (int it = 0; it < 4; ++it) {
      int i = it * 32 + lane;
      int r = i >> 3, c4 = (i & 7) * 4;
      v4bf p = { f2bf(fa[it].x), f2bf(fa[it].y), f2bf(fa[it].z), f2bf(fa[it].w) };
      *(v4bf*)&pA[r * 32 + c4] = p;
    }
    if (transB) {
      #pragma unroll
      for (int it = 0; it < 4; ++it) {
        int i = it * 128 + tid;
        int r = i >> 3, c4 = (i & 7) * 4;
        float m = bmsk[it];
        v4bf p = { f2bf(fb[it].x * m), f2bf(fb[it].y * m),
                   f2bf(fb[it].z * m), f2bf(fb[it].w * m) };
        *(v4bf*)&sB[r * 32 + c4] = p;
      }
    } else {                           // transpose while storing (u16 stores)
      #pragma unroll
      for (int it = 0; it < 4; ++it) {
        int i = it * 128 + tid;
        int kr = i >> 4, c4 = (i & 15) * 4;
        float fv[4] = { fb[it].x, fb[it].y, fb[it].z, fb[it].w };
        #pragma unroll
        for (int j = 0; j < 4; ++j) {
          float v = (n0 + c4 + j < N) ? fv[j] : 0.f;
          sB[(c4 + j) * 32 + kr] = f2bf(v);
        }
      }
    }
    __syncthreads();

    // ====== phase 3: issue ALL fragment ds_load_b128, then WMMA chain =====
    V16U af;
    {
      int mm = lane & 15, kb = (lane >> 4) * 8;
      af.h[0] = *(const v8bf*)&pA[mm * 32 + kb];
      af.h[1] = *(const v8bf*)&pA[mm * 32 + kb + 16];
    }
    V16U bfr[4];
    {
      int nn  = lane & 15;
      int kb2 = (lane >> 4) * 8;
      #pragma unroll
      for (int t = 0; t < 4; ++t) {
        const __bf16* pb = &sB[(t * 16 + nn) * 32 + kb2];
        bfr[t].h[0] = *(const v8bf*)(pb);
        bfr[t].h[1] = *(const v8bf*)(pb + 16);
      }
    }
    #pragma unroll
    for (int t = 0; t < 4; ++t)
      acc[t] = __builtin_amdgcn_wmma_f32_16x16x32_bf16(
          false, af.v, false, bfr[t].v, (short)0, acc[t], false, false);
    __syncthreads();
  }

  // ---- store: VGPR r -> row m0+r / m0+r+8 (lane>=16), col n0+nt*16+(lane&15) ----
  #pragma unroll
  for (int t = 0; t < 4; ++t) {
    int col = n0 + t * 16 + (lane & 15);
    if (col >= N) continue;
    float bv = bias ? bias[col] : 0.f;
    #pragma unroll
    for (int r = 0; r < 8; ++r) {
      int row = m0 + r + ((lane >> 4) << 3);
      float v = acc[t][r] + bv;
      if (Rb)   v += Rb[(long)row * ldc + col];
      if (relu) v = fmaxf(v, 0.f);
      Cb[(long)row * ldc + col] = v;
    }
  }
}

// ---------------- embed: h0[b,t,n,:] = [x*W_in+b | tod | dow | adp] ----------
__global__ void k_embed(const float* __restrict__ x, const float* __restrict__ xcal,
                        const float* __restrict__ Win, const float* __restrict__ bin,
                        const float* __restrict__ todT, const float* __restrict__ dowT,
                        const float* __restrict__ adp, float* __restrict__ h0)
{
  long total = (long)S_ * D_;
  for (long i = (long)blockIdx.x * blockDim.x + threadIdx.x; i < total;
       i += (long)gridDim.x * blockDim.x) {
    int  d  = (int)(i % D_);
    long r  = i / D_;
    int  n  = (int)(r % NV_);
    long bt = r / NV_;
    int  t  = (int)(bt % T_);
    int  b  = (int)(bt / T_);
    float val;
    if (d < 64) {
      val = x[((long)b * NV_ + n) * T_ + t] * Win[d] + bin[d];
    } else if (d < 96) {
      int ti = (int)(xcal[((long)b * T_ + t) * 2 + 1] * 288.f);
      ti = ti < 0 ? 0 : (ti > 287 ? 287 : ti);
      val = todT[ti * 32 + (d - 64)];
    } else if (d < 128) {
      int di = (int)xcal[((long)b * T_ + t) * 2 + 0];
      di = di < 0 ? 0 : (di > 6 ? 6 : di);
      val = dowT[di * 32 + (d - 96)];
    } else {
      val = adp[((long)t * NV_ + n) * 32 + (d - 128)];
    }
    h0[i] = val;
  }
}

// ------------- graph[t,n,:] = softmax(relu(adp[t,n]·adp[t,:]ᵀ)) --------------
__global__ __launch_bounds__(256) void k_graph(const float* __restrict__ adp,
                                               float* __restrict__ graph)
{
  __shared__ float vals[NV_];
  __shared__ float arow[32];
  __shared__ float red[8];
  int t = blockIdx.x / NV_, n = blockIdx.x % NV_;
  const float* an = adp + ((long)t * NV_ + n) * 32;
  if (threadIdx.x < 32) arow[threadIdx.x] = an[threadIdx.x];
  __syncthreads();
  for (int c = threadIdx.x; c < NV_; c += 256) {
    const float* am = adp + ((long)t * NV_ + c) * 32;
    float dot = 0.f;
    #pragma unroll
    for (int k = 0; k < 32; ++k) dot += arow[k] * am[k];
    vals[c] = fmaxf(dot, 0.f);
  }
  __syncthreads();
  float m = -1e30f;
  for (int c = threadIdx.x; c < NV_; c += 256) m = fmaxf(m, vals[c]);
  #pragma unroll
  for (int o = 16; o > 0; o >>= 1) m = fmaxf(m, __shfl_xor(m, o));
  if ((threadIdx.x & 31) == 0) red[threadIdx.x >> 5] = m;
  __syncthreads();
  float bm = red[0];
  #pragma unroll
  for (int i = 1; i < 8; ++i) bm = fmaxf(bm, red[i]);
  __syncthreads();
  float s = 0.f;
  for (int c = threadIdx.x; c < NV_; c += 256) {
    float e = __expf(vals[c] - bm);
    vals[c] = e;
    s += e;
  }
  #pragma unroll
  for (int o = 16; o > 0; o >>= 1) s += __shfl_xor(s, o);
  if ((threadIdx.x & 31) == 0) red[threadIdx.x >> 5] = s;
  __syncthreads();
  float bs = 0.f;
  #pragma unroll
  for (int i = 0; i < 8; ++i) bs += red[i];
  float inv = 1.f / bs;
  float* g = graph + ((long)t * NV_ + n) * NV_;
  for (int c = threadIdx.x; c < NV_; c += 256) g[c] = vals[c] * inv;
}

// --------- per-(row,head) L2 normalize q and k in-place inside qkv -----------
__global__ void k_l2norm_qk(float* __restrict__ qkv)
{
  long total = (long)S_ * 16;                       // 8 heads x {q,k}
  long i = (long)blockIdx.x * blockDim.x + threadIdx.x;
  if (i >= total) return;
  long row = i >> 4;
  int  j   = (int)(i & 15);
  float* p = qkv + row * (3 * D_) + ((j < 8) ? j * HD_ : D_ + (j - 8) * HD_);
  float s = 0.f;
  #pragma unroll
  for (int d = 0; d < HD_; ++d) { float v = p[d]; s += v * v; }
  float inv = 1.f / fmaxf(sqrtf(s), 1e-12f);
  #pragma unroll
  for (int d = 0; d < HD_; ++d) p[d] *= inv;
}

// ------------- spatial Performer attention: block per (b*t, h) ---------------
__global__ __launch_bounds__(512) void k_attn_spatial(const float* __restrict__ qkv,
                                                      float* __restrict__ cat)
{
  int blk = blockIdx.x;
  int h = blk & (H_ - 1);
  int bt = blk >> 3;
  const float* base = qkv + (long)bt * NV_ * (3 * D_);
  __shared__ float kvs[HD_][HD_];
  __shared__ float ksum[HD_];
  int tid = threadIdx.x;
  int m = tid / HD_, d = tid % HD_;
  bool act = tid < HD_ * HD_;
  const int ko = D_ + h * HD_, vo = 2 * D_ + h * HD_, qo = h * HD_;
  float acc = 0.f, accs = 0.f;
  for (int n = 0; n < NV_; ++n) {
    const float* row = base + (long)n * (3 * D_);
    if (act)      acc  += row[ko + m] * row[vo + d];
    if (tid < HD_) accs += row[ko + tid];
  }
  if (act)       kvs[m][d] = acc;
  if (tid < HD_) ksum[tid] = accs;
  __syncthreads();
  {
    int n = tid;                                   // 512 threads == NV_
    const float* row = base + (long)n * (3 * D_);
    float q[HD_];
    #pragma unroll
    for (int i = 0; i < HD_; ++i) q[i] = row[qo + i];
    float den = (float)NV_;
    #pragma unroll
    for (int i = 0; i < HD_; ++i) den += q[i] * ksum[i];
    float* orow = cat + ((long)bt * NV_ + n) * (2 * D_) + h * HD_;
    #pragma unroll
    for (int j = 0; j < HD_; ++j) {
      float num = (float)NV_ * row[vo + j];
      #pragma unroll
      for (int i = 0; i < HD_; ++i) num += q[i] * kvs[i][j];
      orow[j] = num / den;
    }
  }
}

// ------------- temporal Performer attention: wave per (b, n, h) --------------
__global__ void k_attn_temporal(const float* __restrict__ qkv, float* __restrict__ cat)
{
  int unit = blockIdx.x * (blockDim.x >> 5) + (threadIdx.x >> 5);
  int lane = threadIdx.x & 31;
  if (unit >= B_ * NV_ * H_ || lane >= HD_) return;
  int h  = unit & (H_ - 1);
  int bn = unit >> 3;
  int n  = bn & (NV_ - 1);
  int b  = bn >> 9;
  const int qo = h * HD_, ko = D_ + h * HD_, vo = 2 * D_ + h * HD_;
  const int d = lane;
  float kvs[HD_], ksum[HD_];
  #pragma unroll
  for (int i = 0; i < HD_; ++i) { kvs[i] = 0.f; ksum[i] = 0.f; }
  for (int t = 0; t < T_; ++t) {
    const float* row = qkv + (((long)(b * T_ + t) * NV_) + n) * (3 * D_);
    float vd = row[vo + d];
    #pragma unroll
    for (int m = 0; m < HD_; ++m) { float km = row[ko + m]; kvs[m] += km * vd; ksum[m] += km; }
  }
  for (int t = 0; t < T_; ++t) {
    const float* row = qkv + (((long)(b * T_ + t) * NV_) + n) * (3 * D_);
    float num = (float)T_ * row[vo + d];
    float den = (float)T_;
    #pragma unroll
    for (int m = 0; m < HD_; ++m) { float qm = row[qo + m]; num += qm * kvs[m]; den += qm * ksum[m]; }
    cat[(((long)(b * T_ + t) * NV_) + n) * (2 * D_) + D_ + h * HD_ + d] = num / den;
  }
}

// ----------------- x_glo += att * pw * scale (elementwise) -------------------
__global__ void k_combine(float* __restrict__ xglo, const float* __restrict__ att,
                          const float* __restrict__ pw, float scale, long n)
{
  for (long i = (long)blockIdx.x * blockDim.x + threadIdx.x; i < n;
       i += (long)gridDim.x * blockDim.x)
    xglo[i] += att[i] * pw[i] * scale;
}

// -------- layernorm over D=160: out = LN(alpha*a (+res)) * g + b -------------
__global__ void k_layernorm(const float* __restrict__ a, const float* __restrict__ res,
                            float alpha, const float* __restrict__ g,
                            const float* __restrict__ be, float* __restrict__ out, int rows)
{
  int row  = blockIdx.x * (blockDim.x >> 5) + (threadIdx.x >> 5);
  int lane = threadIdx.x & 31;
  if (row >= rows) return;
  const float* pa = a + (long)row * D_;
  const float* pr = res ? res + (long)row * D_ : nullptr;
  float v[5];
  #pragma unroll
  for (int i = 0; i < 5; ++i) {
    int idx = i * 32 + lane;
    v[i] = alpha * pa[idx] + (pr ? pr[idx] : 0.f);
  }
  float s = v[0] + v[1] + v[2] + v[3] + v[4];
  #pragma unroll
  for (int o = 16; o > 0; o >>= 1) s += __shfl_xor(s, o);
  float mean = s * (1.f / D_);
  float q = 0.f;
  #pragma unroll
  for (int i = 0; i < 5; ++i) { float dd = v[i] - mean; q += dd * dd; }
  #pragma unroll
  for (int o = 16; o > 0; o >>= 1) q += __shfl_xor(q, o);
  float rstd = rsqrtf(q * (1.f / D_) + 1e-5f);
  float* po = out + (long)row * D_;
  #pragma unroll
  for (int i = 0; i < 5; ++i) {
    int idx = i * 32 + lane;
    po[idx] = (v[i] - mean) * rstd * g[idx] + be[idx];
  }
}

// ------------- (B,T,N,D) -> (B,N,T*D) physical transpose ---------------------
__global__ void k_to_bn(const float* __restrict__ in, float* __restrict__ out)
{
  long total = (long)S_ * D_;
  for (long i = (long)blockIdx.x * blockDim.x + threadIdx.x; i < total;
       i += (long)gridDim.x * blockDim.x) {
    int  d  = (int)(i % D_);
    long r  = i / D_;
    int  n  = (int)(r % NV_);
    long bt = r / NV_;
    int  t  = (int)(bt % T_);
    int  b  = (int)(bt / T_);
    out[(((long)b * NV_ + n) * T_ + t) * D_ + d] = in[i];
  }
}

// ------------- (B,N,12) -> (B,12,N,1) output permute -------------------------
__global__ void k_final(const float* __restrict__ t12, float* __restrict__ out)
{
  int total = BN_ * 12;
  int i = blockIdx.x * blockDim.x + threadIdx.x;
  if (i >= total) return;
  int s  = i % 12;
  int bn = i / 12;
  int n  = bn & (NV_ - 1);
  int b  = bn >> 9;
  out[((long)(b * 12 + s) * NV_) + n] = t12[i];
}

// ---------------------------------------------------------------------------
static inline void gemm(hipStream_t st,
                        const float* A, int lda, long aBatch, int aMod,
                        const float* Bm, int ldb, long bBatch, int transB,
                        const float* bias, const float* res,
                        float* C, int ldc, long cBatch,
                        int M, int N, int K, int relu, int batches)
{
  dim3 grid((N + 63) / 64, (M + 63) / 64, batches);
  k_gemm<<<grid, 128, 0, st>>>(A, lda, aBatch, aMod, Bm, ldb, bBatch, transB,
                               bias, res, C, ldc, cBatch, M, N, K, relu);
}

extern "C" void kernel_launch(void* const* d_in, const int* in_sizes, int n_in,
                              void* d_out, int out_size, void* d_ws, size_t ws_size,
                              hipStream_t stream)
{
  (void)in_sizes; (void)n_in; (void)out_size; (void)ws_size;
  // params in setup_inputs() insertion order
  const float* x     = (const float*)d_in[0];
  const float* xcal  = (const float*)d_in[1];
  const float* W_in  = (const float*)d_in[2];
  const float* b_in  = (const float*)d_in[3];
  const float* todT  = (const float*)d_in[4];
  const float* dowT  = (const float*)d_in[5];
  const float* adp   = (const float*)d_in[6];
  const float* W_tp  = (const float*)d_in[7];
  const float* b_tp  = (const float*)d_in[8];
  const float* Wqkv[2] = {(const float*)d_in[9],  (const float*)d_in[12]};
  const float* Wout[2] = {(const float*)d_in[10], (const float*)d_in[13]};
  const float* bout[2] = {(const float*)d_in[11], (const float*)d_in[14]};
  const float* pwW[2]  = {(const float*)d_in[15], (const float*)d_in[17]};
  const float* pwB[2]  = {(const float*)d_in[16], (const float*)d_in[18]};
  const float* fc1W  = (const float*)d_in[19];
  const float* fc1b  = (const float*)d_in[20];
  const float* fc2W  = (const float*)d_in[21];
  const float* fc2b  = (const float*)d_in[22];
  const float* ln1g  = (const float*)d_in[23];
  const float* ln1b  = (const float*)d_in[24];
  const float* ln2g  = (const float*)d_in[25];
  const float* ln2b  = (const float*)d_in[26];
  const float* W_enc = (const float*)d_in[27];
  const float* b_enc = (const float*)d_in[28];
  const float* encW1[3] = {(const float*)d_in[29], (const float*)d_in[33], (const float*)d_in[37]};
  const float* encB1[3] = {(const float*)d_in[30], (const float*)d_in[34], (const float*)d_in[38]};
  const float* encW2[3] = {(const float*)d_in[31], (const float*)d_in[35], (const float*)d_in[39]};
  const float* encB2[3] = {(const float*)d_in[32], (const float*)d_in[36], (const float*)d_in[40]};
  const float* WoutF = (const float*)d_in[41];
  const float* boutF = (const float*)d_in[42];
  float* out = (float*)d_out;

  // ---- workspace carving (floats). Requires ~12 * 126MB = 1.51 GB. ----
  float* ws = (float*)d_ws;
  const long SD = (long)S_ * D_;
  float* h0   = ws + 0 * SD;
  float* h    = ws + 1 * SD;
  float* x1   = ws + 2 * SD;
  float* qkv  = ws + 3 * SD;   // 3 slots (S x 480)
  float* cat  = ws + 6 * SD;   // 2 slots (S x 320)
  float* att0 = ws + 8 * SD;
  float* att1 = ws + 9 * SD;
  float* pwb  = ws + 10 * SD;
  float* xglo = ws + 11 * SD;
  // phase-B aliases (earlier buffers dead by then)
  float* xln   = h0;            // ln1 output
  float* mhid  = h;             // S x 320 (slots 1-2)
  float* t2    = qkv;           // mlp output (slot 3)
  float* x2    = qkv + SD;      // ln2 output (slot 4)
  float* hbn   = qkv + 2 * SD;  // (B,N,T*D)  (slot 5)
  float* graphb = cat;          // T*512*512 = 3.1M floats; dead before cat written
  float* he    = cat;           // BN x 160
  float* ehid  = cat + 4L * 1024 * 1024;   // BN x 320
  float* outT  = cat + 12L * 1024 * 1024;  // BN x 12

  const long elemSD = SD;
  const int  EB = 256;

  // 1) embed -> h0
  k_embed<<<(int)((elemSD + EB - 1) / EB), EB, 0, stream>>>(x, xcal, W_in, b_in, todT, dowT, adp, h0);

  // 2) temporal proj: h = h0 @ W_tp^T + b_tp
  gemm(stream, h0, D_, 0, 1, W_tp, D_, 0, 1, b_tp, nullptr, h, D_, 0, S_, D_, D_, 0, 1);

  // 3) graph = softmax(relu(adp adp^T)) per t
  k_graph<<<T_ * NV_, 256, 0, stream>>>(adp, graphb);

  // 4) x1[b,t] = graph[t] @ h[b,t]  (batched 512x512 * 512x160)
  gemm(stream, graphb, NV_, (long)NV_ * NV_, T_,
       h, D_, (long)NV_ * D_, 0, nullptr, nullptr,
       x1, D_, (long)NV_ * D_, NV_, D_, NV_, 0, B_ * T_);

  // 5) x_glo = h  (then accumulate attention branches)
  hipMemcpyAsync(xglo, h, (size_t)SD * sizeof(float), hipMemcpyDeviceToDevice, stream);

  const float scales[2] = {1.0f, 0.01f};
  const float* zin[2] = {h, x1};
  float* attb[2] = {att0, att1};
  for (int it = 0; it < 2; ++it) {
    // qkv = z @ W_qkv^T
    gemm(stream, zin[it], D_, 0, 1, Wqkv[it], D_, 0, 1, nullptr, nullptr,
         qkv, 3 * D_, 0, S_, 3 * D_, D_, 0, 1);
    // l2-normalize q,k per head in-place
    {
      long tot = (long)S_ * 16;
      k_l2norm_qk<<<(int)((tot + 255) / 256), 256, 0, stream>>>(qkv);
    }
    // spatial & temporal Performer attention -> cat halves
    k_attn_spatial<<<B_ * T_ * H_, 512, 0, stream>>>(qkv, cat);
    k_attn_temporal<<<(B_ * NV_ * H_) / 8, 256, 0, stream>>>(qkv, cat);
    // att = cat @ W_out^T + b_out
    gemm(stream, cat, 2 * D_, 0, 1, Wout[it], 2 * D_, 0, 1, bout[it], nullptr,
         attb[it], D_, 0, S_, D_, 2 * D_, 0, 1);
    // pw = c @ pwW^T + pwb   (c = h for it 0, att0 for it 1)
    const float* c_in = (it == 0) ? h : att0;
    gemm(stream, c_in, D_, 0, 1, pwW[it], D_, 0, 1, pwB[it], nullptr,
         pwb, D_, 0, S_, D_, D_, 0, 1);
    // x_glo += att * pw * scale
    k_combine<<<(int)((elemSD + 255) / 256), 256, 0, stream>>>(xglo, attb[it], pwb, scales[it], elemSD);
  }

  // 6) x = LN(2 * x_glo)
  k_layernorm<<<(S_ + 7) / 8, 256, 0, stream>>>(xglo, nullptr, 2.0f, ln1g, ln1b, xln, S_);

  // 7) MLP + LN2: x2 = LN(x + fc2(relu(fc1(x))))
  gemm(stream, xln, D_, 0, 1, fc1W, D_, 0, 1, fc1b, nullptr, mhid, HID_, 0, S_, HID_, D_, 1, 1);
  gemm(stream, mhid, HID_, 0, 1, fc2W, HID_, 0, 1, fc2b, nullptr, t2, D_, 0, S_, D_, HID_, 0, 1);
  k_layernorm<<<(S_ + 7) / 8, 256, 0, stream>>>(t2, xln, 1.0f, ln2g, ln2b, x2, S_);

  // 8) transpose to (B,N,T*D)
  k_to_bn<<<(int)((elemSD + 255) / 256), 256, 0, stream>>>(x2, hbn);

  // 9) he = hbn @ W_enc^T + b_enc
  gemm(stream, hbn, TD_, 0, 1, W_enc, TD_, 0, 1, b_enc, nullptr, he, D_, 0, BN_, D_, TD_, 0, 1);

  // 10) residual encoder MLPs x3: he = mlp(he) + he
  for (int i = 0; i < 3; ++i) {
    gemm(stream, he, D_, 0, 1, encW1[i], D_, 0, 1, encB1[i], nullptr,
         ehid, HID_, 0, BN_, HID_, D_, 1, 1);
    gemm(stream, ehid, HID_, 0, 1, encW2[i], HID_, 0, 1, encB2[i], he,
         he, D_, 0, BN_, D_, HID_, 0, 1);
  }

  // 11) outT = he @ W_out^T + b_out  (BN x 12), then permute to (B,12,N,1)
  gemm(stream, he, D_, 0, 1, WoutF, D_, 0, 1, boutF, nullptr, outT, 12, 0, BN_, 12, D_, 0, 1);
  k_final<<<(BN_ * 12 + 255) / 256, 256, 0, stream>>>(outT, out);
}
